// VectorQuantizer_45397804319287
// MI455X (gfx1250) — matrix operand order repaired and checked by reading, not compiled, not run on hardware
//
#include <hip/hip_runtime.h>
#include <stdint.h>

typedef float  v8f   __attribute__((ext_vector_type(8)));
typedef int    v8i   __attribute__((ext_vector_type(8)));
typedef __bf16 v16bf __attribute__((ext_vector_type(16)));

#define NUM_EMB 8192
#define EDIM 256
#define NROWS 8192            // 8 * 32 * 32
#define BHW 1024              // 32 * 32
#define ZB_STRIDE (EDIM*BHW)  // 262144 floats per batch image
#define LOSS_OFF 2097152
#define IDX_OFF  2097153

#define CCHUNK 32             // K per LDS stage (one bf16 WMMA K-slab)
#define LDS_STRIDE 20         // dwords per row: 16 slots + 4 pad (16B-aligned quads)

// Round f32 to bf16 (RNE) via integer ops; returns the 32-bit float bits of hi.
__device__ __forceinline__ unsigned int bf16_rne_bits(float x) {
    unsigned int u = __float_as_uint(x);
    u += 0x7FFFu + ((u >> 16) & 1u);
    return u & 0xFFFF0000u;     // bf16 value as f32 bit pattern
}

// Split two f32 into packed bf16 hi pair + bf16 lo (residual) pair.
__device__ __forceinline__ void split2(float x0, float x1,
                                       unsigned int& hi, unsigned int& lo) {
    unsigned int h0 = bf16_rne_bits(x0);
    unsigned int h1 = bf16_rne_bits(x1);
    float r0 = x0 - __uint_as_float(h0);
    float r1 = x1 - __uint_as_float(h1);
    unsigned int l0 = bf16_rne_bits(r0);
    unsigned int l1 = bf16_rne_bits(r1);
    hi = (h0 >> 16) | h1;       // even K in [15:0], odd K in [31:16]
    lo = (l0 >> 16) | l1;
}

// Load 8 consecutive dwords (32B, 16B-aligned) from LDS as one fragment.
__device__ __forceinline__ v8i lds_load8(const unsigned int* p) {
    union { uint4 q[2]; v8i v; } u;
    u.q[0] = *(const uint4*)(p);
    u.q[1] = *(const uint4*)(p + 4);
    return u.v;
}

// ---------------- Kernel 1: codebook norms (exact f32) + atomic-min init ----------------
__global__ __launch_bounds__(256)
void vq_prep(const float* __restrict__ emb, float* __restrict__ norms,
             unsigned long long* __restrict__ packed) {
    int k = blockIdx.x;
    int t = threadIdx.x;
    float e = emb[(long long)k * EDIM + t];
    __shared__ float red[256];
    red[t] = e * e;
    __syncthreads();
    for (int s = 128; s > 0; s >>= 1) {
        if (t < s) red[t] += red[t + s];
        __syncthreads();
    }
    if (t == 0) {
        norms[k] = red[0];
        packed[k] = 0xFFFFFFFFFFFFFFFFULL;
    }
}

// ---------------- Kernel 2: split-bf16 WMMA GEMM + fused row argmin ----------------
// Block computes a 128(rows) x 128(codes) tile of S = X * E^T with
// S ~= Xh*Eh + Xl*Eh + Xh*El accumulated in f32 (error ~2^-18 relative).
// Score minimized: ||e_k||^2 - 2 * x.e_k  (||x||^2 constant per row, dropped).
//
// LDS layout puts each wave's WMMA fragment in ONE contiguous 32B region:
//   A rows store K-pair slots with bit2<->bit3 swapped so the ISA A-fragment
//   (pairs {4h..4h+3, 8+4h..11+4h}) is slots [8h, 8h+8) in VGPR order.
//   B rows use the same bit2<->bit3 swap, so the B-fragment is also slots [8h, 8h+8).
__global__ __launch_bounds__(256)
void vq_gemm_argmin(const float* __restrict__ z, const float* __restrict__ emb,
                    const float* __restrict__ norms,
                    unsigned long long* __restrict__ packed) {
    __shared__ unsigned int Ah[128][LDS_STRIDE];  // [n][slot]
    __shared__ unsigned int Al[128][LDS_STRIDE];
    __shared__ unsigned int Bh[128][LDS_STRIDE];  // [k][slot]
    __shared__ unsigned int Bl[128][LDS_STRIDE];
    __shared__ float rv[2][128];
    __shared__ int   ri[2][128];

    const int t     = threadIdx.x;
    const int lane  = t & 31;
    const int wid   = t >> 5;
    const int wm    = wid & 3;    // 4 waves over rows -> 32 rows each
    const int wn    = wid >> 2;   // 2 waves over cols -> 64 cols each
    const int m16   = lane & 15;
    const int khalf = lane >> 4;  // lane half selects K sub-ranges

    const int n0  = blockIdx.x * 128;
    const int k0  = blockIdx.y * 128;
    const int b   = n0 >> 10;
    const int hw0 = n0 & 1023;
    const long long zbase = (long long)b * ZB_STRIDE + hw0;

    const v8f zero = {0.f, 0.f, 0.f, 0.f, 0.f, 0.f, 0.f, 0.f};
    v8f acc[2][4];
#pragma unroll
    for (int i = 0; i < 2; ++i)
#pragma unroll
        for (int j = 0; j < 4; ++j) acc[i][j] = zero;

    for (int cc = 0; cc < EDIM; cc += CCHUNK) {
        // Stage+split A chunk: 16 c-pairs x 128 n. Coalesced 128-float runs per c.
#pragma unroll
        for (int i = 0; i < 8; ++i) {
            int idx = i * 256 + t;        // 0..2047
            int c2  = idx >> 7;           // 0..15
            int nl  = idx & 127;
            float x0 = z[zbase + (long long)(cc + 2 * c2) * BHW + nl];
            float x1 = z[zbase + (long long)(cc + 2 * c2 + 1) * BHW + nl];
            unsigned int hi, lo;
            split2(x0, x1, hi, lo);
            int slot = (c2 & 3) | ((c2 & 4) << 1) | ((c2 & 8) >> 1);  // swap bit2<->bit3
            Ah[nl][slot] = hi;
            Al[nl][slot] = lo;
        }
        // Stage+split B chunk: 128 k x 16 c-pairs. One float2 (8B) per thread.
#pragma unroll
        for (int i = 0; i < 8; ++i) {
            int idx = i * 256 + t;
            int kl  = idx >> 4;           // 0..127
            int c2  = idx & 15;
            float2 e2 = *(const float2*)&emb[(long long)(k0 + kl) * EDIM + cc + 2 * c2];
            unsigned int hi, lo;
            split2(e2.x, e2.y, hi, lo);
            int slot = (c2 & 3) | ((c2 & 4) << 1) | ((c2 & 8) >> 1);  // swap bit2<->bit3
            Bh[kl][slot] = hi;
            Bl[kl][slot] = lo;
        }
        if (cc + CCHUNK < EDIM) {
            __builtin_prefetch(&z[zbase + (long long)(cc + CCHUNK) * BHW + (t & 127)], 0, 3);
            __builtin_prefetch(&emb[(long long)(k0 + (t >> 1)) * EDIM + cc + CCHUNK], 0, 3);
        }
        __syncthreads();

        // A fragments: one contiguous 32B LDS region per (tile, hi/lo).
        v16bf ahf[2], alf[2];
#pragma unroll
        for (int i = 0; i < 2; ++i) {
            int r = wm * 32 + i * 16 + m16;
            ahf[i] = __builtin_bit_cast(v16bf, lds_load8(&Ah[r][8 * khalf]));
            alf[i] = __builtin_bit_cast(v16bf, lds_load8(&Al[r][8 * khalf]));
        }
#pragma unroll
        for (int j = 0; j < 4; ++j) {
            int cidx = wn * 64 + j * 16 + m16;
            v16bf bhf = __builtin_bit_cast(v16bf, lds_load8(&Bh[cidx][8 * khalf]));
            v16bf blf = __builtin_bit_cast(v16bf, lds_load8(&Bl[cidx][8 * khalf]));
#pragma unroll
            for (int i = 0; i < 2; ++i) {
                acc[i][j] = __builtin_amdgcn_wmma_f32_16x16x32_bf16(
                    false, ahf[i], false, blf, (short)0, acc[i][j], false, false);
                acc[i][j] = __builtin_amdgcn_wmma_f32_16x16x32_bf16(
                    false, alf[i], false, bhf, (short)0, acc[i][j], false, false);
                acc[i][j] = __builtin_amdgcn_wmma_f32_16x16x32_bf16(
                    false, ahf[i], false, bhf, (short)0, acc[i][j], false, false);
            }
        }
        __syncthreads();
    }

    // Per-lane adjustment terms (||e||^2 for this lane's columns).
    float adjc[4];
#pragma unroll
    for (int j = 0; j < 4; ++j) adjc[j] = norms[k0 + wn * 64 + j * 16 + m16];

    // Row-wise argmin. D layout: VGPR v, lane L -> row v + 8*(L>>4), col L&15.
#pragma unroll
    for (int i = 0; i < 2; ++i) {
#pragma unroll
        for (int v = 0; v < 8; ++v) {
            float bv = adjc[0] - 2.0f * acc[i][0][v];
            int   bc = k0 + wn * 64 + m16;
#pragma unroll
            for (int j = 1; j < 4; ++j) {
                float tv = adjc[j] - 2.0f * acc[i][j][v];
                int   tc = k0 + wn * 64 + j * 16 + m16;
                if (tv < bv) { bv = tv; bc = tc; }   // strict <: ties keep lower index
            }
#pragma unroll
            for (int m = 8; m >= 1; m >>= 1) {
                float ov = __shfl_xor(bv, m, 32);
                int   oc = __shfl_xor(bc, m, 32);
                if (ov < bv || (ov == bv && oc < bc)) { bv = ov; bc = oc; }
            }
            if (m16 == 0) {
                int row = wm * 32 + i * 16 + v + 8 * khalf;
                rv[wn][row] = bv;
                ri[wn][row] = bc;
            }
        }
    }
    __syncthreads();

    if (t < 128) {
        float bv = rv[0][t]; int bc = ri[0][t];
        float ov = rv[1][t]; int oc = ri[1][t];
        if (ov < bv || (ov == bv && oc < bc)) { bv = ov; bc = oc; }
        // Monotone float->uint key; pack (key, col) so u64 atomicMin gives
        // min-value then min-index — bitwise deterministic across replays.
        unsigned int u = __float_as_uint(bv);
        unsigned int key = (u & 0x80000000u) ? ~u : (u | 0x80000000u);
        unsigned long long pk = ((unsigned long long)key << 32) | (unsigned int)bc;
        atomicMin(&packed[n0 + t], pk);
    }
}

// ---------------- Kernel 3: gather + straight-through output + loss partials ----------------
__global__ __launch_bounds__(256)
void vq_finalize(const float* __restrict__ z, const float* __restrict__ emb,
                 const unsigned long long* __restrict__ packed,
                 float* __restrict__ out, float* __restrict__ partial) {
    int bc = blockIdx.x;          // 0..2047 : (batch, channel)
    int b = bc >> 8;
    int c = bc & 255;
    int t = threadIdx.x;
    long long base = (long long)b * ZB_STRIDE + (long long)c * BHW;
    float s = 0.f;
    for (int hw = t; hw < BHW; hw += 256) {
        int idx = (int)(unsigned int)(packed[b * BHW + hw] & 0xFFFFFFFFULL);
        float zv = z[base + hw];
        float qv = emb[(long long)idx * EDIM + c];
        float d = qv - zv;
        out[base + hw] = zv + d;   // quantized_st = z + (q - z), same rounding as ref
        s += d * d;
    }
    __shared__ float red[256];
    red[t] = s;
    __syncthreads();
    for (int k = 128; k > 0; k >>= 1) {
        if (t < k) red[t] += red[t + k];
        __syncthreads();
    }
    if (t == 0) partial[bc] = red[0];
}

// ---------------- Kernel 4a: indices output ----------------
__global__ __launch_bounds__(256)
void vq_indices(const unsigned long long* __restrict__ packed, float* __restrict__ out) {
    int n = blockIdx.x * 256 + threadIdx.x;
    out[IDX_OFF + n] = (float)(unsigned int)(packed[n] & 0xFFFFFFFFULL);
}

// ---------------- Kernel 4b: loss reduction (fixed order, deterministic) ----------------
__global__ __launch_bounds__(256)
void vq_loss(const float* __restrict__ partial, float* __restrict__ out) {
    int t = threadIdx.x;
    float s = 0.f;
    for (int i = t; i < 2048; i += 256) s += partial[i];
    __shared__ float red[256];
    red[t] = s;
    __syncthreads();
    for (int k = 128; k > 0; k >>= 1) {
        if (t < k) red[t] += red[t + k];
        __syncthreads();
    }
    // loss = codebook + 0.25*commitment = 1.25 * mean((q-z)^2)
    if (t == 0) out[LOSS_OFF] = 1.25f * red[0] / 2097152.0f;
}

extern "C" void kernel_launch(void* const* d_in, const int* in_sizes, int n_in,
                              void* d_out, int out_size, void* d_ws, size_t ws_size,
                              hipStream_t stream) {
    (void)in_sizes; (void)n_in; (void)out_size; (void)ws_size;
    const float* z   = (const float*)d_in[0];
    const float* emb = (const float*)d_in[1];
    float* out = (float*)d_out;

    // Workspace layout: [0,64K) u64 packed[8192] | [64K,96K) norms | [96K,+8K) partials
    unsigned long long* packed = (unsigned long long*)d_ws;
    float* norms   = (float*)((char*)d_ws + 65536);
    float* partial = (float*)((char*)d_ws + 98304);

    vq_prep<<<NUM_EMB, 256, 0, stream>>>(emb, norms, packed);
    dim3 grid(64, 64);
    vq_gemm_argmin<<<grid, 256, 0, stream>>>(z, emb, norms, packed);
    vq_finalize<<<2048, 256, 0, stream>>>(z, emb, packed, out, partial);
    vq_indices<<<NROWS / 256, 256, 0, stream>>>(packed, out);
    vq_loss<<<1, 256, 0, stream>>>(partial, out);
}
